// c_4_75445395521982
// MI455X (gfx1250) — compile-verified, matrix operand order checked by reading, not run on hardware
//
#include <hip/hip_runtime.h>
#include <hip/hip_bf16.h>
#include <math.h>

typedef __attribute__((ext_vector_type(16))) _Float16 v16h;
typedef __attribute__((ext_vector_type(8)))  float    v8f;
typedef __attribute__((address_space(3)))    float    lds_f32;

// ---------------------------------------------------------------------------
// Kernel A: fused conv1 -> relu(maxpool3) -> conv2 -> relu(maxpool3) -> linear
// One workgroup per image (640 images). Double-buffered band streaming via
// GLOBAL_LOAD_ASYNC_TO_LDS_B128 (ASYNCcnt), overlapping HBM reads with VALU.
// ---------------------------------------------------------------------------
__global__ __launch_bounds__(256) void conv_front_kernel(
    const float* __restrict__ nodes,   // (640,3,224,224)
    const float* __restrict__ pos,     // (640,6)
    const float* __restrict__ c1w, const float* __restrict__ c1b,
    const float* __restrict__ c2w, const float* __restrict__ c2b,
    const float* __restrict__ linw, const float* __restrict__ linb,
    float* __restrict__ nodes_all)     // (640,12)
{
    __shared__ float sIn[2][3 * 13 * 224]; // double-buffered input bands
    __shared__ float sP1[3 * 37 * 37];     // pooled conv1 output
    __shared__ float sC2[18 * 18];         // conv2 output
    __shared__ float sPool[36];            // pooled conv2 output

    const int img = blockIdx.x;
    const int tid = threadIdx.x;
    const size_t imgBase = (size_t)img * 3 * 224 * 224;

    // conv1 weights: uniform addresses -> scalar loads
    float w1[3][3][3][3];
#pragma unroll
    for (int o = 0; o < 3; ++o)
#pragma unroll
        for (int i = 0; i < 3; ++i)
#pragma unroll
            for (int y = 0; y < 3; ++y)
#pragma unroll
                for (int x = 0; x < 3; ++x)
                    w1[o][i][y][x] = c1w[((o * 3 + i) * 3 + y) * 3 + x];
    const float b1_0 = c1b[0], b1_1 = c1b[1], b1_2 = c1b[2];

    // async-issue one band (13 or 7 rows x 3ch x 224 cols) into sIn[band&1]
    auto issue_band = [&](int band) {
        const int pr0  = band * 2;
        const int prc  = (pr0 + 2 <= 37) ? 2 : 1;
        const int rows = 6 * prc + 1;
        const int ir0  = 6 * pr0;
        const int buf  = band & 1;
        const int nvec = 3 * rows * 56;           // 56 float4 per row
        for (int v = tid; v < nvec; v += 256) {
            const int ch  = v / (rows * 56);
            const int rem = v - ch * rows * 56;
            const int r   = rem / 56;
            const int c4  = (rem - r * 56) * 4;
            const float* src = nodes + imgBase + ((size_t)ch * 224 + (ir0 + r)) * 224 + c4;
            const unsigned dst =
                (unsigned)(size_t)(lds_f32*)&sIn[buf][(ch * 13 + r) * 224 + c4];
            // direct global -> LDS async copy (16B/lane), tracked by ASYNCcnt
            asm volatile("global_load_async_to_lds_b128 %0, %1, off"
                         :: "v"(dst), "v"(src) : "memory");
        }
    };

    // 37 pooled rows in bands of 2 (19 bands, last band = 1 row), pipelined.
    issue_band(0);
    for (int band = 0; band < 19; ++band) {
        // wait for this band's async copies, then make them visible to the WG
        asm volatile("s_wait_asynccnt 0x0" ::: "memory");
        __syncthreads();

        if (band + 1 < 19) issue_band(band + 1);
        if (band + 2 < 19 && tid < 64) {  // warm L2 for band+2
            __builtin_prefetch(nodes + imgBase + (size_t)(12 * (band + 2)) * 224 + tid * 16, 0, 1);
        }

        const int pr0  = band * 2;
        const int prc  = (pr0 + 2 <= 37) ? 2 : 1;
        const float* bandBuf = &sIn[band & 1][0];

        // each thread: all 3 output channels of one pooled conv1 pixel
        for (int p = tid; p < prc * 37; p += 256) {
            const int prl = p / 37;
            const int pc  = p - prl * 37;
            float m0 = -1e30f, m1 = -1e30f, m2 = -1e30f;
#pragma unroll
            for (int cr = 0; cr < 3; ++cr) {
#pragma unroll
                for (int cc = 0; cc < 3; ++cc) {
                    float a0 = b1_0, a1 = b1_1, a2 = b1_2;
                    const int rbase = 6 * prl + 2 * cr;
                    const int cbase = 6 * pc + 2 * cc;
#pragma unroll
                    for (int ic = 0; ic < 3; ++ic) {
#pragma unroll
                        for (int ky = 0; ky < 3; ++ky) {
                            const float* rp = &bandBuf[(ic * 13 + rbase + ky) * 224 + cbase];
#pragma unroll
                            for (int kx = 0; kx < 3; ++kx) {
                                const float vv = rp[kx];
                                a0 += vv * w1[0][ic][ky][kx];
                                a1 += vv * w1[1][ic][ky][kx];
                                a2 += vv * w1[2][ic][ky][kx];
                            }
                        }
                    }
                    m0 = fmaxf(m0, a0); m1 = fmaxf(m1, a1); m2 = fmaxf(m2, a2);
                }
            }
            const int pr = pr0 + prl;
            sP1[(0 * 37 + pr) * 37 + pc] = fmaxf(m0, 0.f);
            sP1[(1 * 37 + pr) * 37 + pc] = fmaxf(m1, 0.f);
            sP1[(2 * 37 + pr) * 37 + pc] = fmaxf(m2, 0.f);
        }
        __syncthreads();   // all waves done with sIn[band&1] before reuse
    }

    // conv2 (3ch -> 1ch, 3x3, stride 2): 18x18 outputs
    float w2[27];
#pragma unroll
    for (int k = 0; k < 27; ++k) w2[k] = c2w[k];
    const float b2 = c2b[0];
    for (int p = tid; p < 324; p += 256) {
        const int i = p / 18, j = p - (p / 18) * 18;
        float a = b2;
#pragma unroll
        for (int ic = 0; ic < 3; ++ic)
#pragma unroll
            for (int ky = 0; ky < 3; ++ky)
#pragma unroll
                for (int kx = 0; kx < 3; ++kx)
                    a += sP1[(ic * 37 + 2 * i + ky) * 37 + 2 * j + kx] * w2[(ic * 3 + ky) * 3 + kx];
        sC2[p] = a;
    }
    __syncthreads();

    // relu(maxpool3): 6x6 = 36 features
    if (tid < 36) {
        const int pi = tid / 6, pj = tid - (tid / 6) * 6;
        float m = -1e30f;
#pragma unroll
        for (int ky = 0; ky < 3; ++ky)
#pragma unroll
            for (int kx = 0; kx < 3; ++kx)
                m = fmaxf(m, sC2[(3 * pi + ky) * 18 + 3 * pj + kx]);
        sPool[tid] = fmaxf(m, 0.f);
    }
    __syncthreads();

    // linear 36->6 and concat with pos -> nodes_all (12 per image)
    if (tid < 12) {
        if (tid < 6) {
            float s = linb[tid];
#pragma unroll
            for (int k = 0; k < 36; ++k) s += sPool[k] * linw[tid * 36 + k];
            nodes_all[img * 12 + tid] = s;
        } else {
            nodes_all[img * 12 + tid] = pos[img * 6 + (tid - 6)];
        }
    }
}

// ---------------------------------------------------------------------------
// Kernel B: graph head. One workgroup per batch element (32 blocks).
// Produces h = update.reshape(n,240).
// ---------------------------------------------------------------------------
__global__ __launch_bounds__(128) void graph_kernel(
    const float* __restrict__ nodes_all, // (32,5,4,12) flattened
    const float* __restrict__ attmap,    // (32,5,4,4)
    const float* __restrict__ wfc_w, const float* __restrict__ wfc_b,
    const float* __restrict__ frame_w, const float* __restrict__ frame_b,
    const float* __restrict__ last_w, const float* __restrict__ last_b,
    float* __restrict__ hout)            // (32,240)
{
    __shared__ float na[240];
    __shared__ float sp[20];
    __shared__ float tp[20];
    __shared__ float msg[240];
    const int n   = blockIdx.x;
    const int tid = threadIdx.x;

    for (int i = tid; i < 240; i += 128) na[i] = nodes_all[n * 240 + i];
    __syncthreads();

    if (tid < 20) {
        float s = 0.f, t = 0.f;
#pragma unroll
        for (int k = 0; k < 12; ++k) {
            const float x = na[tid * 12 + k];
            s += x * wfc_w[k];
            t += x * wfc_w[12 + k];
        }
        sp[tid] = s; tp[tid] = t;
    }
    for (int i = tid; i < 240; i += 128) {
        const int fs = i / 12, d = i - fs * 12;
        float s = frame_b[d];
#pragma unroll
        for (int k = 0; k < 12; ++k) s += na[fs * 12 + k] * frame_w[d * 12 + k];
        msg[i] = s;
    }
    __syncthreads();

    const float wb = wfc_b[0];
    for (int i = tid; i < 240; i += 128) {
        const int f = i / 48;
        const int r = i - f * 48;
        const int t = r / 12;
        const int d = r - t * 12;
        float agg = 0.f;
#pragma unroll
        for (int s = 0; s < 4; ++s) {
            const float x  = sp[f * 4 + s] + tp[f * 4 + t] + wb;
            const float lw = 1.f / (1.f + __expf(-x)) + attmap[((n * 5 + f) * 4 + s) * 4 + t];
            agg += lw * msg[(f * 4 + s) * 12 + d];
        }
        if (f >= 1) {
            float l = last_b[d];
#pragma unroll
            for (int k = 0; k < 12; ++k)
                l += na[((f - 1) * 4 + t) * 12 + k] * last_w[d * 12 + k];
            agg += l;
        }
        hout[n * 240 + i] = agg;
    }
}

// ---------------------------------------------------------------------------
// Kernel C: MLP head via f16 WMMA with f32 accumulate.
//   fc1: (32x240)@(240x120)  -> M=32 (2 tiles), N pad 128 (8), K pad 256 (8)
//   fc2: (32x120)@(120x60)   -> N pad 64 (4), K pad 128 (4)
//   fc3: (32x60)@(60x6)      -> scalar
// ---------------------------------------------------------------------------
__global__ __launch_bounds__(256) void mlp_head_kernel(
    const float* __restrict__ h,     // (32,240)
    const float* __restrict__ fc1_w, const float* __restrict__ fc1_b,
    const float* __restrict__ fc2_w, const float* __restrict__ fc2_b,
    const float* __restrict__ fc3_w, const float* __restrict__ fc3_b,
    float* __restrict__ out)         // (32,6)
{
    __shared__ _Float16 A1[32][256];   // activations, K padded 240->256
    __shared__ _Float16 Bbuf[32][128]; // weight K-chunk (shared fc1/fc2)
    __shared__ _Float16 A2[32][128];   // fc1 output (relu), K padded 120->128
    __shared__ float    act2[32][64];  // fc2 output (relu), N padded 60->64

    const int tid  = threadIdx.x;
    const int lane = tid & 31;
    const int wave = tid >> 5;
    const int m16  = lane & 15;
    const int hi   = lane >> 4;       // upper half-wave handles K+8 / M+8

    // stage A1 = f16(h), zero-padded
    for (int i = tid; i < 32 * 256; i += 256) {
        const int m = i >> 8, k = i & 255;
        A1[m][k] = (k < 240) ? (_Float16)h[m * 240 + k] : (_Float16)0.f;
    }

    // ---- fc1: wave w owns tiles (mt=0,nt=w) and (mt=1,nt=w) ----
    v8f acc0 = {};
    v8f acc1 = {};
    const int nt = wave;
    for (int kt = 0; kt < 8; ++kt) {
        __syncthreads();
        for (int i = tid; i < 32 * 128; i += 256) {
            const int kk = i >> 7, nn = i & 127;
            const int kg = kt * 32 + kk;
            Bbuf[kk][nn] = (nn < 120 && kg < 240) ? (_Float16)fc1_w[nn * 240 + kg]
                                                  : (_Float16)0.f;
        }
        __syncthreads();
        v16h bfrag, a0, a1;
#pragma unroll
        for (int v = 0; v < 8; ++v) {
            const int kk = ((v >= 4) ? 16 : 0) + hi * 8 + (v & 3) * 2;
            bfrag[2 * v]     = Bbuf[kk][nt * 16 + m16];
            bfrag[2 * v + 1] = Bbuf[kk + 1][nt * 16 + m16];
            const int kg = kt * 32 + kk;
            a0[2 * v]     = A1[m16][kg];
            a0[2 * v + 1] = A1[m16][kg + 1];
            a1[2 * v]     = A1[16 + m16][kg];
            a1[2 * v + 1] = A1[16 + m16][kg + 1];
        }
        acc0 = __builtin_amdgcn_wmma_f32_16x16x32_f16(false, a0, false, bfrag,
                                                      (short)0, acc0, false, false);
        acc1 = __builtin_amdgcn_wmma_f32_16x16x32_f16(false, a1, false, bfrag,
                                                      (short)0, acc1, false, false);
    }
    __syncthreads();
    // bias + relu -> A2 (f16), padded columns stay zero
    {
        const int col  = nt * 16 + m16;
        const float bb = (col < 120) ? fc1_b[col] : 0.f;
#pragma unroll
        for (int r = 0; r < 8; ++r) {
            const int row = r + hi * 8;           // C/D layout: VGPR r, half-wave hi
            float v0 = acc0[r] + bb; v0 = v0 > 0.f ? v0 : 0.f;
            float v1 = acc1[r] + bb; v1 = v1 > 0.f ? v1 : 0.f;
            A2[row][col]      = (_Float16)v0;
            A2[16 + row][col] = (_Float16)v1;
        }
    }

    // ---- fc2: 8 tiles, wave w owns tile (mt=w>>2, nt=w&3) ----
    v8f acc2 = {};
    const int mt2 = wave >> 2, nt2 = wave & 3;
    for (int kt = 0; kt < 4; ++kt) {
        __syncthreads();
        for (int i = tid; i < 32 * 64; i += 256) {
            const int kk = i >> 6, nn = i & 63;
            const int kg = kt * 32 + kk;
            Bbuf[kk][nn] = (nn < 60 && kg < 120) ? (_Float16)fc2_w[nn * 120 + kg]
                                                 : (_Float16)0.f;
        }
        __syncthreads();
        v16h bfrag, afrag;
#pragma unroll
        for (int v = 0; v < 8; ++v) {
            const int kk = ((v >= 4) ? 16 : 0) + hi * 8 + (v & 3) * 2;
            bfrag[2 * v]     = Bbuf[kk][nt2 * 16 + m16];
            bfrag[2 * v + 1] = Bbuf[kk + 1][nt2 * 16 + m16];
            const int kg = kt * 32 + kk;
            afrag[2 * v]     = A2[mt2 * 16 + m16][kg];
            afrag[2 * v + 1] = A2[mt2 * 16 + m16][kg + 1];
        }
        acc2 = __builtin_amdgcn_wmma_f32_16x16x32_f16(false, afrag, false, bfrag,
                                                      (short)0, acc2, false, false);
    }
    __syncthreads();
    {
        const int col  = nt2 * 16 + m16;
        const float bb = (col < 60) ? fc2_b[col] : 0.f;
#pragma unroll
        for (int r = 0; r < 8; ++r) {
            const int row = mt2 * 16 + r + hi * 8;
            const float v = acc2[r] + bb;
            act2[row][col] = v > 0.f ? v : 0.f;
        }
    }
    __syncthreads();

    // ---- fc3 (60 -> 6), scalar ----
    if (tid < 192) {
        const int n = tid / 6, o = tid - (tid / 6) * 6;
        float s = fc3_b[o];
#pragma unroll
        for (int k = 0; k < 60; ++k) s += act2[n][k] * fc3_w[o * 60 + k];
        out[n * 6 + o] = s;
    }
}

// ---------------------------------------------------------------------------
// Host launcher
// ---------------------------------------------------------------------------
extern "C" void kernel_launch(void* const* d_in, const int* in_sizes, int n_in,
                              void* d_out, int out_size, void* d_ws, size_t ws_size,
                              hipStream_t stream) {
    const float* nodes   = (const float*)d_in[0];
    const float* pos     = (const float*)d_in[1];
    const float* attmap  = (const float*)d_in[2];
    // d_in[3] = depths (unused by the reference)
    const float* conv1_w = (const float*)d_in[4];
    const float* conv1_b = (const float*)d_in[5];
    const float* conv2_w = (const float*)d_in[6];
    const float* conv2_b = (const float*)d_in[7];
    const float* lin_w   = (const float*)d_in[8];
    const float* lin_b   = (const float*)d_in[9];
    const float* wfc_w   = (const float*)d_in[10];
    const float* wfc_b   = (const float*)d_in[11];
    const float* frame_w = (const float*)d_in[12];
    const float* frame_b = (const float*)d_in[13];
    const float* last_w  = (const float*)d_in[14];
    const float* last_b  = (const float*)d_in[15];
    const float* fc1_w   = (const float*)d_in[16];
    const float* fc1_b   = (const float*)d_in[17];
    const float* fc2_w   = (const float*)d_in[18];
    const float* fc2_b   = (const float*)d_in[19];
    const float* fc3_w   = (const float*)d_in[20];
    const float* fc3_b   = (const float*)d_in[21];

    float* ws        = (float*)d_ws;
    float* nodes_all = ws;          // 640*12  = 7680 floats
    float* hbuf      = ws + 7680;   // 32*240  = 7680 floats

    conv_front_kernel<<<640, 256, 0, stream>>>(
        nodes, pos, conv1_w, conv1_b, conv2_w, conv2_b, lin_w, lin_b, nodes_all);

    graph_kernel<<<32, 128, 0, stream>>>(
        nodes_all, attmap, wfc_w, wfc_b, frame_w, frame_b, last_w, last_b, hbuf);

    mlp_head_kernel<<<1, 256, 0, stream>>>(
        hbuf, fc1_w, fc1_b, fc2_w, fc2_b, fc3_w, fc3_b, (float*)d_out);
}
